// SNNLanguageModel_44315472560254
// MI455X (gfx1250) — matrix operand order repaired and checked by reading, not compiled, not run on hardware
//
#include <hip/hip_runtime.h>
#include <stddef.h>

// ---------------------------------------------------------------------------
// SNN language model forward for MI455X (gfx1250).
// Strategy: HBM-bound problem (weights re-read K*S times) -> convert all
// weights + embedding to bf16 once per launch (halves bytes), run every GEMM
// on v_wmma_f32_16x16x32_bf16 with fp32 accumulate. Sequential recurrence is
// driven by a host loop of small kernels (graph-capture safe, stream only).
// ---------------------------------------------------------------------------

typedef __bf16 bf16_t;
typedef __attribute__((ext_vector_type(16))) __bf16 v16bf;
typedef __attribute__((ext_vector_type(8)))  __bf16 v8bf;
typedef __attribute__((ext_vector_type(8)))  float  v8f;

#define BB     4
#define SS     16
#define DD     1024
#define KK     8
#define NBB    4
#define DNN    2048
#define VOCAB  151936

__device__ __forceinline__ float sigmoidf_(float x) {
  return 1.0f / (1.0f + __expf(-x));
}

// ------------------------------- conversions -------------------------------

__global__ void cvt_f32_bf16(const float* __restrict__ src,
                             bf16_t* __restrict__ dst, size_t n) {
  size_t i = (size_t)blockIdx.x * blockDim.x + threadIdx.x;
  size_t stride = (size_t)gridDim.x * blockDim.x;
  for (; i < n; i += stride) dst[i] = (bf16_t)src[i];
}

__global__ void zero_u32(unsigned* __restrict__ p, size_t n) {
  size_t i = (size_t)blockIdx.x * blockDim.x + threadIdx.x;
  size_t stride = (size_t)gridDim.x * blockDim.x;
  for (; i < n; i += stride) p[i] = 0u;
}

// ------------------------------ WMMA GEMM ----------------------------------
// Y(4 x Dout) = A(16 x DinA, rows 4..15 zero) @ W(Dout x DinA)^T
//             [+ A2(16 x DinB) @ W2(Dout x DinB)^T] + bias
// One wave per 16-wide N tile; K loop in steps of 32 via
// v_wmma_f32_16x16x32_bf16. Layouts used here (contiguous k order, same for A and B):
//   A (16-bit 16x32): lane<16 -> K 0..15, lane>=16 -> K 16..31
//   B (16-bit 32x16): N = lane&15, lane<16 -> K 0..15, lane>=16 -> K 16..31
//   C/D (32-bit 16x16): lanes 0..15 hold N, VGPR r = row M=r (M 0..7)

struct GemmSeg {
  const bf16_t* A;
  const bf16_t* W;
  const bf16_t* A2;   // optional (nullptr if unused)
  const bf16_t* W2;
  const float*  bias; // optional
  float*        Y;    // row stride ldY
  int DinA, DinB, Dout, ldY;
};
struct GemmArgs { GemmSeg s[4]; };

__device__ __forceinline__ v8f wmma_k32(const bf16_t* __restrict__ Arow,
                                        const bf16_t* __restrict__ Wrow,
                                        int Din, int aoff, int boff, v8f acc) {
  for (int kk = 0; kk < Din; kk += 32) {
    v8bf a_lo = *(const v8bf*)(Arow + kk + aoff);
    v8bf a_hi = *(const v8bf*)(Arow + kk + 8 + aoff);
    v16bf a = __builtin_shufflevector(a_lo, a_hi,
        0,1,2,3,4,5,6,7,8,9,10,11,12,13,14,15);
    v8bf b_lo = *(const v8bf*)(Wrow + kk + boff);
    v8bf b_hi = *(const v8bf*)(Wrow + kk + boff + 8);
    v16bf b = __builtin_shufflevector(b_lo, b_hi,
        0,1,2,3,4,5,6,7,8,9,10,11,12,13,14,15);
    __builtin_prefetch(Wrow + kk + 256, 0, 1);   // global_prefetch_b8
    acc = __builtin_amdgcn_wmma_f32_16x16x32_bf16(
        false, a, false, b, (short)0, acc, false, false);
  }
  return acc;
}

__global__ __launch_bounds__(256) void gemm_bf16_wmma(GemmArgs args) {
  const GemmSeg seg = args.s[blockIdx.y];
  const int lane   = threadIdx.x & 31;
  const int waveId = threadIdx.x >> 5;
  const int tile   = blockIdx.x * (blockDim.x >> 5) + waveId;
  const int n0     = tile * 16;
  if (n0 >= seg.Dout) return;               // wave-uniform, EXEC stays full

  const int mrow = lane & 15;
  const int aoff = (lane < 16) ? 0 : 16;
  const int nl   = lane & 15;
  const int boff = (lane < 16) ? 0 : 16;

  v8f acc = {};
  {
    const bf16_t* Arow = seg.A + (size_t)mrow * seg.DinA;
    const bf16_t* Wrow = seg.W + (size_t)(n0 + nl) * seg.DinA;
    acc = wmma_k32(Arow, Wrow, seg.DinA, aoff, boff, acc);
  }
  if (seg.A2) {
    const bf16_t* Arow = seg.A2 + (size_t)mrow * seg.DinB;
    const bf16_t* Wrow = seg.W2 + (size_t)(n0 + nl) * seg.DinB;
    acc = wmma_k32(Arow, Wrow, seg.DinB, aoff, boff, acc);
  }

  if (lane < 16) {
    const float bias = seg.bias ? seg.bias[n0 + lane] : 0.0f;
#pragma unroll
    for (int m = 0; m < BB; ++m)
      seg.Y[(size_t)m * seg.ldY + n0 + lane] = acc[m] + bias;
  }
}

// ----------------------------- elementwise ---------------------------------

__global__ void embed_gather(const int* __restrict__ ids,
                             const float* __restrict__ embed,
                             bf16_t* __restrict__ emb16, int s) {
  int idx = blockIdx.x * blockDim.x + threadIdx.x;
  if (idx >= BB * DD) return;
  int b = idx >> 10, d = idx & (DD - 1);
  int tok = ids[b * SS + s];
  emb16[idx] = (bf16_t)embed[(size_t)tok * DD + d];
}

// res = sigmoid(hpre); decoded = 0
__global__ void post_enc(const float* __restrict__ hpre,
                         float* __restrict__ res,
                         float* __restrict__ decoded) {
  int idx = blockIdx.x * blockDim.x + threadIdx.x;
  if (idx >= BB * DD) return;
  res[idx] = sigmoidf_(hpre[idx]);
  decoded[idx] = 0.0f;
}

// bit = (res>=0.5); res = (res - 0.5*bit)*2; x = bit
__global__ void encode_frame(float* __restrict__ res,
                             bf16_t* __restrict__ xD16) {
  int idx = blockIdx.x * blockDim.x + threadIdx.x;
  if (idx >= BB * DD) return;
  float r = res[idx];
  float bit = (r >= 0.5f) ? 1.0f : 0.0f;
  res[idx] = (r - 0.5f * bit) * 2.0f;
  xD16[idx] = (bf16_t)bit;
}

// membrane update: beta/alpha/vth gates, soft reset, emit spikes
__global__ void state_update(const float* __restrict__ gb,
                             const float* __restrict__ ga,
                             const float* __restrict__ gt,
                             const float* __restrict__ gi,
                             float* __restrict__ V,
                             bf16_t* __restrict__ spDN16) {
  int idx = blockIdx.x * blockDim.x + threadIdx.x;
  if (idx >= BB * DNN) return;
  float beta  = sigmoidf_(gb[idx]);
  float alpha = sigmoidf_(ga[idx]);
  float t = gt[idx];
  float sp_arg = (t > 20.0f) ? t : log1pf(expf(t));   // softplus
  float vth = 0.1f + sp_arg;
  float Vn = beta * V[idx] + alpha * gi[idx];
  float sp = (Vn - vth >= 0.0f) ? 1.0f : 0.0f;
  V[idx] = Vn - sp * vth;
  spDN16[idx] = (bf16_t)sp;
}

// PLIF output neuron: cur = sigmoid(gate)*o + skip; u update; hard reset
__global__ void out_update(const float* __restrict__ gatepre,
                           const float* __restrict__ obuf,
                           const float* __restrict__ skipb,
                           float* __restrict__ u,
                           const float* __restrict__ plif_w,
                           const float* __restrict__ out_vth, int blk,
                           bf16_t* __restrict__ xD16,
                           float* __restrict__ decoded,
                           float bitw, int accum) {
  int idx = blockIdx.x * blockDim.x + threadIdx.x;
  if (idx >= BB * DD) return;
  float plif = sigmoidf_(plif_w[blk]);
  float cur = sigmoidf_(gatepre[idx]) * obuf[idx] + skipb[idx];
  float un = u[idx] + (cur - u[idx]) * plif;
  float so = (un - out_vth[blk] >= 0.0f) ? 1.0f : 0.0f;
  u[idx] = (1.0f - so) * un;
  xD16[idx] = (bf16_t)so;
  if (accum) decoded[idx] += bitw * so;
}

__global__ __launch_bounds__(256) void rmsnorm_to_bf16(
    const float* __restrict__ hhpre, const float* __restrict__ norm_w,
    bf16_t* __restrict__ hh16) {
  __shared__ float red[256];
  int b = blockIdx.x;
  float s = 0.0f;
  for (int d = threadIdx.x; d < DD; d += 256) {
    float v = hhpre[b * DD + d];
    s += v * v;
  }
  red[threadIdx.x] = s;
  __syncthreads();
  for (int off = 128; off > 0; off >>= 1) {
    if (threadIdx.x < off) red[threadIdx.x] += red[threadIdx.x + off];
    __syncthreads();
  }
  float inv = rsqrtf(red[0] / (float)DD + 1e-6f);
  for (int d = threadIdx.x; d < DD; d += 256)
    hh16[b * DD + d] = (bf16_t)(norm_w[d] * hhpre[b * DD + d] * inv);
}

// ------------------------------- host side ---------------------------------

static inline size_t align256(size_t x) { return (x + 255) & ~(size_t)255; }

extern "C" void kernel_launch(void* const* d_in, const int* in_sizes, int n_in,
                              void* d_out, int out_size, void* d_ws, size_t ws_size,
                              hipStream_t stream) {
  (void)in_sizes; (void)n_in; (void)out_size; (void)ws_size;

  // -------- inputs (setup_inputs order) --------
  const int*   token_ids = (const int*)  d_in[0];
  const float* embed     = (const float*)d_in[1];
  const float* norm_w    = (const float*)d_in[2];
  const float* enc_W     = (const float*)d_in[3];
  const float* enc_b     = (const float*)d_in[4];
  const float* dec_W     = (const float*)d_in[5];
  const float* dec_b     = (const float*)d_in[6];
  const float* Win       = (const float*)d_in[7];
  const float* WbX       = (const float*)d_in[8];
  const float* WaX       = (const float*)d_in[9];
  const float* WtX       = (const float*)d_in[10];
  const float* WbV       = (const float*)d_in[11];
  const float* WaV       = (const float*)d_in[12];
  const float* WtV       = (const float*)d_in[13];
  const float* b_beta    = (const float*)d_in[14];
  const float* b_alpha   = (const float*)d_in[15];
  const float* b_th      = (const float*)d_in[16];
  const float* Wgate     = (const float*)d_in[17];
  const float* Wskip     = (const float*)d_in[18];
  const float* Wout      = (const float*)d_in[19];
  const float* plif_w    = (const float*)d_in[20];
  const float* out_vth   = (const float*)d_in[21];
  float* out = (float*)d_out;

  // -------- workspace layout --------
  char* ws = (char*)d_ws;
  size_t cur = 0;
  auto allocF = [&](size_t elems) -> float* {
    float* p = (float*)(ws + cur); cur = align256(cur + elems * 4); return p;
  };
  auto allocH = [&](size_t elems) -> bf16_t* {
    bf16_t* p = (bf16_t*)(ws + cur); cur = align256(cur + elems * 2); return p;
  };

  // activation region (zeroed every call; padded bf16 rows 4..15 stay 0)
  float* hpre    = allocF(BB * DD);
  float* res     = allocF(BB * DD);
  float* decoded = allocF(BB * DD);
  float* hhpre   = allocF(BB * DD);
  float* gb      = allocF(BB * DNN);
  float* ga      = allocF(BB * DNN);
  float* gt      = allocF(BB * DNN);
  float* gi      = allocF(BB * DNN);
  float* gatepre = allocF(BB * DD);
  float* skipb   = allocF(BB * DD);
  float* obuf    = allocF(BB * DD);
  float* Vbuf    = allocF((size_t)NBB * BB * DNN);
  float* ubuf    = allocF((size_t)NBB * BB * DD);
  bf16_t* emb16  = allocH(16 * DD);
  bf16_t* xD16   = allocH(16 * DD);
  bf16_t* dec16  = allocH(16 * DD);
  bf16_t* hh16   = allocH(16 * DD);
  bf16_t* VDN16  = allocH(16 * DNN);
  bf16_t* spDN16 = allocH(16 * DNN);
  const size_t zero_bytes = cur;

  // bf16 weight copies (halve HBM traffic; weights re-read K*S times)
  bf16_t* emb_bf  = allocH((size_t)VOCAB * DD);
  bf16_t* encW_bf = allocH((size_t)DD * DD);
  bf16_t* decW_bf = allocH((size_t)DD * DD);
  bf16_t* Win_bf  = allocH((size_t)NBB * DNN * DD);
  bf16_t* WbX_bf  = allocH((size_t)NBB * DNN * DD);
  bf16_t* WaX_bf  = allocH((size_t)NBB * DNN * DD);
  bf16_t* WtX_bf  = allocH((size_t)NBB * DNN * DD);
  bf16_t* WbV_bf  = allocH((size_t)NBB * DNN * DNN);
  bf16_t* WaV_bf  = allocH((size_t)NBB * DNN * DNN);
  bf16_t* WtV_bf  = allocH((size_t)NBB * DNN * DNN);
  bf16_t* Wg_bf   = allocH((size_t)NBB * DD * DD);
  bf16_t* Wsk_bf  = allocH((size_t)NBB * DD * DD);
  bf16_t* Wo_bf   = allocH((size_t)NBB * DD * DNN);

  auto cvt = [&](const float* src, bf16_t* dst, size_t n) {
    size_t blocks = (n + 255) / 256;
    if (blocks > 4096) blocks = 4096;
    hipLaunchKernelGGL(cvt_f32_bf16, dim3((unsigned)blocks), dim3(256), 0,
                       stream, src, dst, n);
  };

  // zero activation region + persistent V/u state
  hipLaunchKernelGGL(zero_u32, dim3(512), dim3(256), 0, stream,
                     (unsigned*)ws, zero_bytes / 4);

  cvt(embed, emb_bf, (size_t)VOCAB * DD);
  cvt(enc_W, encW_bf, (size_t)DD * DD);
  cvt(dec_W, decW_bf, (size_t)DD * DD);
  cvt(Win, Win_bf, (size_t)NBB * DNN * DD);
  cvt(WbX, WbX_bf, (size_t)NBB * DNN * DD);
  cvt(WaX, WaX_bf, (size_t)NBB * DNN * DD);
  cvt(WtX, WtX_bf, (size_t)NBB * DNN * DD);
  cvt(WbV, WbV_bf, (size_t)NBB * DNN * DNN);
  cvt(WaV, WaV_bf, (size_t)NBB * DNN * DNN);
  cvt(WtV, WtV_bf, (size_t)NBB * DNN * DNN);
  cvt(Wgate, Wg_bf, (size_t)NBB * DD * DD);
  cvt(Wskip, Wsk_bf, (size_t)NBB * DD * DD);
  cvt(Wout, Wo_bf, (size_t)NBB * DD * DNN);

  auto launch_gemm = [&](const GemmArgs& gargs, int nseg, int maxDout) {
    int tiles  = (maxDout + 15) / 16;
    int blocks = (tiles + 7) / 8;                 // 8 waves / 256-thread block
    hipLaunchKernelGGL(gemm_bf16_wmma, dim3(blocks, nseg), dim3(256), 0,
                       stream, gargs);
  };

  const int ewD  = (BB * DD  + 255) / 256;
  const int ewDN = (BB * DNN + 255) / 256;

  for (int s = 0; s < SS; ++s) {
    // embedding -> encoder -> residual for bit encoding
    hipLaunchKernelGGL(embed_gather, dim3(ewD), dim3(256), 0, stream,
                       token_ids, embed, emb16, s);
    {
      GemmArgs gA = {};
      gA.s[0] = { emb16, encW_bf, nullptr, nullptr, enc_b, hpre,
                  DD, 0, DD, DD };
      launch_gemm(gA, 1, DD);
    }
    hipLaunchKernelGGL(post_enc, dim3(ewD), dim3(256), 0, stream,
                       hpre, res, decoded);

    for (int k = 0; k < KK; ++k) {
      hipLaunchKernelGGL(encode_frame, dim3(ewD), dim3(256), 0, stream,
                         res, xD16);
      const float bitw = 1.0f / (float)(2 << k);   // 2^-(k+1), MSB first

      for (int i = 0; i < NBB; ++i) {
        const size_t oXD = (size_t)i * DNN * DD;   // (DN x D) block offset
        const size_t oVV = (size_t)i * DNN * DNN;  // (DN x DN)
        const size_t oDD = (size_t)i * DD * DD;    // (D x D)
        const size_t oON = (size_t)i * DD * DNN;   // (D x DN)
        float* Vi = Vbuf + (size_t)i * BB * DNN;
        float* ui = ubuf + (size_t)i * BB * DD;

        // V (fp32 state) -> bf16 tile input for the V-side GEMMs
        hipLaunchKernelGGL(cvt_f32_bf16, dim3(ewDN), dim3(256), 0, stream,
                           Vi, VDN16, (size_t)BB * DNN);

        // fused 4-segment GEMM: beta/alpha/vth (x + V sides) and Win (x side)
        GemmArgs g1 = {};
        g1.s[0] = { xD16, WbX_bf + oXD, VDN16, WbV_bf + oVV,
                    b_beta  + (size_t)i * DNN, gb, DD, DNN, DNN, DNN };
        g1.s[1] = { xD16, WaX_bf + oXD, VDN16, WaV_bf + oVV,
                    b_alpha + (size_t)i * DNN, ga, DD, DNN, DNN, DNN };
        g1.s[2] = { xD16, WtX_bf + oXD, VDN16, WtV_bf + oVV,
                    b_th    + (size_t)i * DNN, gt, DD, DNN, DNN, DNN };
        g1.s[3] = { xD16, Win_bf + oXD, nullptr, nullptr,
                    nullptr, gi, DD, 0, DNN, DNN };
        launch_gemm(g1, 4, DNN);

        hipLaunchKernelGGL(state_update, dim3(ewDN), dim3(256), 0, stream,
                           gb, ga, gt, gi, Vi, spDN16);

        // fused 3-segment GEMM: gate(x), skip(x), out(spikes)
        GemmArgs g2 = {};
        g2.s[0] = { xD16,   Wg_bf  + oDD, nullptr, nullptr, nullptr,
                    gatepre, DD, 0, DD, DD };
        g2.s[1] = { xD16,   Wsk_bf + oDD, nullptr, nullptr, nullptr,
                    skipb,   DD, 0, DD, DD };
        g2.s[2] = { spDN16, Wo_bf  + oON, nullptr, nullptr, nullptr,
                    obuf,    DNN, 0, DD, DD };
        launch_gemm(g2, 3, DD);

        hipLaunchKernelGGL(out_update, dim3(ewD), dim3(256), 0, stream,
                           gatepre, obuf, skipb, ui, plif_w, out_vth, i,
                           xD16, decoded, bitw, (i == NBB - 1) ? 1 : 0);
      }
    }

    // decode -> RMSNorm -> tied LM head into d_out[b][s][:]
    hipLaunchKernelGGL(cvt_f32_bf16, dim3(ewD), dim3(256), 0, stream,
                       decoded, dec16, (size_t)BB * DD);
    {
      GemmArgs gD = {};
      gD.s[0] = { dec16, decW_bf, nullptr, nullptr, dec_b, hhpre,
                  DD, 0, DD, DD };
      launch_gemm(gD, 1, DD);
    }
    hipLaunchKernelGGL(rmsnorm_to_bf16, dim3(BB), dim3(256), 0, stream,
                       hhpre, norm_w, hh16);
    {
      GemmArgs gL = {};
      gL.s[0] = { hh16, emb_bf, nullptr, nullptr, nullptr,
                  out + (size_t)s * VOCAB, DD, 0, VOCAB, SS * VOCAB };
      launch_gemm(gL, 1, VOCAB);
    }
  }
}
